// Encoder_18545668784632
// MI455X (gfx1250) — matrix operand order repaired and checked by reading, not compiled, not run on hardware
//
#include <hip/hip_runtime.h>
#include <stdint.h>

// Problem dims (fixed by reference)
#define B_   64
#define T_   1024
#define IN_  512
#define U_   512
#define G3_  1536   // 3*U

typedef __attribute__((ext_vector_type(16))) __bf16       bf16x16;
typedef __attribute__((ext_vector_type(8)))  float        f32x8;
typedef __attribute__((ext_vector_type(4)))  unsigned int u32x4;

union Frag { u32x4 q[2]; bf16x16 v; };   // 32 bytes: one WMMA 16-bit A/B fragment per lane

__device__ __forceinline__ unsigned short f2bf(float f) {
  unsigned int u = __float_as_uint(f);
  u += 0x7FFFu + ((u >> 16) & 1u);          // round-to-nearest-even
  return (unsigned short)(u >> 16);
}
__device__ __forceinline__ float bf2f(unsigned short h) {
  return __uint_as_float(((unsigned int)h) << 16);
}
__device__ __forceinline__ float sigmoidf_(float x) {
  return 1.0f / (1.0f + __expf(-x));
}

// ---------------------------------------------------------------------------
// Elementwise f32 -> bf16
// ---------------------------------------------------------------------------
__global__ void k_f32_to_bf16(const float* __restrict__ in,
                              unsigned short* __restrict__ out, long long n) {
  long long i = (long long)blockIdx.x * blockDim.x + threadIdx.x;
  long long stride = (long long)gridDim.x * blockDim.x;
  for (; i < n; i += stride) out[i] = f2bf(in[i]);
}

// Transpose [K,N] f32 -> [N,K] bf16  (puts GEMM B-operand in K-contiguous form)
__global__ void k_transpose_bf16(const float* __restrict__ in,
                                 unsigned short* __restrict__ out, int K, int N) {
  long long total = (long long)K * N;
  long long i = (long long)blockIdx.x * blockDim.x + threadIdx.x;
  long long stride = (long long)gridDim.x * blockDim.x;
  for (; i < total; i += stride) {
    int k = (int)(i / N);
    int n = (int)(i % N);
    out[(long long)n * K + k] = f2bf(in[i]);
  }
}

// ---------------------------------------------------------------------------
// Input-projection GEMM: C[M,N] = A[M,K](bf16) * Bt[N,K](bf16)^T + bias[N]
// Block = 8 waves; wave w owns 16 M-rows; block tile = 128(M) x 64(N).
// K is a template constant so the pipelined K-loop fully unrolls: the
// double-buffer indices constant-fold and all fragments stay in VGPRs.
// ---------------------------------------------------------------------------
template <int K>
__global__ __launch_bounds__(256) void k_gemm_bf16(
    const unsigned short* __restrict__ A,
    const unsigned short* __restrict__ Bt,
    const float* __restrict__ bias,
    float* __restrict__ C,
    int M, int N) {
  __shared__ unsigned short Bs[2][64][40];   // 80B row stride: 16B aligned, bank-spread

  const int lane = threadIdx.x & 31;
  const int wave = threadIdx.x >> 5;
  const int hi   = lane >> 4;      // 0: lanes 0-15, 1: lanes 16-31
  const int nlo  = lane & 15;

  const int m0 = blockIdx.y * 128 + wave * 16;   // M is a multiple of 128
  const int n0 = blockIdx.x * 64;

  // cooperative B-tile copy mapping: 256 threads x 16B = 64x32 bf16 tile
  const int crow = threadIdx.x >> 2;
  const int cseg = threadIdx.x & 3;
  const unsigned short* csrc = Bt + (long long)(n0 + crow) * K + cseg * 8;

  const unsigned short* arow = A + (long long)(m0 + nlo) * K;

  f32x8 zf8 = {};
  f32x8 acc[4];
  acc[0] = zf8; acc[1] = zf8; acc[2] = zf8; acc[3] = zf8;

  auto loadA = [&](int k0, Frag& f) {
    f.q[0] = *(const u32x4*)(arow + k0 + hi * 8);
    f.q[1] = *(const u32x4*)(arow + k0 + 16 + hi * 8);
  };

  // prologue: stage K-step 0
  u32x4 creg = *(const u32x4*)(csrc);
  *(u32x4*)&Bs[0][crow][cseg * 8] = creg;
  Frag a[2];
  loadA(0, a[0]);
  __syncthreads();

  constexpr int nk = K >> 5;
#pragma unroll
  for (int kk = 0; kk < nk; ++kk) {
    const int cur = kk & 1;
    u32x4 nreg;
    if (kk + 1 < nk) {                       // fetch next tile while computing
      nreg = *(const u32x4*)(csrc + (kk + 1) * 32);
      loadA((kk + 1) * 32, a[cur ^ 1]);
    }
#pragma unroll
    for (int nt = 0; nt < 4; ++nt) {
      Frag b;                                // lane nlo = column, table-order K from LDS
      const unsigned short* brow = &Bs[cur][nt * 16 + nlo][hi * 8];
      b.q[0] = *(const u32x4*)(brow);
      b.q[1] = *(const u32x4*)(brow + 16);
      acc[nt] = __builtin_amdgcn_wmma_f32_16x16x32_bf16(
          false, a[cur].v, false, b.v, (short)0, acc[nt], false, false);
    }
    if (kk + 1 < nk) {
      *(u32x4*)&Bs[cur ^ 1][crow][cseg * 8] = nreg;
      __syncthreads();                       // single barrier per K-step
    }
  }

#pragma unroll
  for (int nt = 0; nt < 4; ++nt) {
#pragma unroll
    for (int e = 0; e < 8; ++e) {
      int m = m0 + e + hi * 8;               // C layout: VGPR e, lane -> (M,N)
      int n = n0 + nt * 16 + nlo;
      C[(long long)m * N + n] = acc[nt][e] + bias[n];
    }
  }
}

// ---------------------------------------------------------------------------
// GRU recurrence over T steps. Grid = 4 blocks (16-row batch tiles), 1024 thr
// (32 waves, 8 per SIMD for latency hiding on the serial path).
// Hidden state (16x512 bf16) lives in LDS; R streamed bf16 from L2 each step.
// Wave w owns unit-tile w: computes z/r/candidate columns for units
// [16w,16w+16) so all gate coupling is wave-local. Fragments for K-step k+1
// are loaded (register double-buffer) while the 3 WMMAs of step k execute.
// The K-tile order is rotated by (t & 15) so R-fragment addresses are
// t-dependent: this blocks LICM from hoisting 1.5KB/lane of loop-invariant
// loads out of the T-loop (which previously spilled all R frags to scratch),
// and staggers the L2 access phase across timesteps.
// ---------------------------------------------------------------------------
__global__ __launch_bounds__(1024) void k_gru_recur(
    const unsigned short* __restrict__ Rt,   // [1536,512] bf16 (R transposed)
    const float* __restrict__ bh,            // [1536] recurrent bias
    const float* __restrict__ xp,            // [B,T,1536] f32 (x*W + bi)
    unsigned short* __restrict__ ys_bf,      // [B,T,512] bf16 out (or null)
    float* __restrict__ ys_f32,              // [B,T,512] f32 out (or null)
    float* __restrict__ h_final) {           // [B,512] f32
  __shared__ unsigned short hbuf[16][520];   // 1040B row stride: 16B aligned, bank-staggered

  const int lane = threadIdx.x & 31;
  const int wave = threadIdx.x >> 5;         // 0..31 -> unit tile
  const int hi   = lane >> 4;
  const int nlo  = lane & 15;
  const int b0   = blockIdx.x * 16;
  const int ub   = wave * 16;                // unit-tile base column

  for (int i = threadIdx.x; i < 16 * 520; i += blockDim.x)
    (&hbuf[0][0])[i] = 0;                    // h0 = 0
  __syncthreads();

  f32x8 zf8 = {};
  float hnew[8];
#pragma unroll
  for (int e = 0; e < 8; ++e) hnew[e] = 0.0f;

  const unsigned short* hrow = &hbuf[nlo][0];
  auto load_frags = [&](int k0, Frag& a, Frag* b) {
    a.q[0] = *(const u32x4*)(hrow + k0 + hi * 8);
    a.q[1] = *(const u32x4*)(hrow + k0 + 16 + hi * 8);
#pragma unroll
    for (int g = 0; g < 3; ++g) {            // z, r, candidate rows of Rt
      const unsigned short* brow =
          Rt + (long long)(g * U_ + ub + nlo) * U_ + k0 + hi * 8;
      b[g].q[0] = *(const u32x4*)(brow);
      b[g].q[1] = *(const u32x4*)(brow + 16);
    }
  };

  for (int t = 0; t < T_; ++t) {
    f32x8 acc[3];
    acc[0] = zf8; acc[1] = zf8; acc[2] = zf8;

    // t-dependent K-tile rotation: sum over all 16 tiles is order-invariant.
    const int krot = t & 15;

    Frag afr[2];
    Frag bfr[2][3];
    load_frags((krot & 15) * 32, afr[0], bfr[0]);
#pragma unroll
    for (int kk = 0; kk < 16; ++kk) {
      const int cur = kk & 1;
      if (kk < 15) load_frags(((kk + 1 + krot) & 15) * 32, afr[cur ^ 1], bfr[cur ^ 1]);
#pragma unroll
      for (int g = 0; g < 3; ++g)
        acc[g] = __builtin_amdgcn_wmma_f32_16x16x32_bf16(
            false, afr[cur].v, false, bfr[cur][g].v, (short)0, acc[g],
            false, false);
    }

    // Gates (all three accs share the identical C lane mapping)
    const int ucol = ub + nlo;
#pragma unroll
    for (int e = 0; e < 8; ++e) {
      const int m = e + hi * 8;
      const float* xrow = xp + ((long long)(b0 + m) * T_ + t) * G3_;
      float z    = sigmoidf_(xrow[ucol]          + acc[0][e] + bh[ucol]);
      float r    = sigmoidf_(xrow[U_ + ucol]     + acc[1][e] + bh[U_ + ucol]);
      float innh =                                  acc[2][e] + bh[2 * U_ + ucol];
      float hh   = tanhf(xrow[2 * U_ + ucol] + r * innh);
      float hold = bf2f(hbuf[m][ucol]);
      hnew[e] = z * hold + (1.0f - z) * hh;
    }
    __syncthreads();                          // all reads of h done

#pragma unroll
    for (int e = 0; e < 8; ++e) {
      const int m = e + hi * 8;
      unsigned short hb = f2bf(hnew[e]);
      hbuf[m][ucol] = hb;
      long long oidx = ((long long)(b0 + m) * T_ + t) * U_ + ucol;
      if (ys_bf)  ys_bf[oidx]  = hb;
      if (ys_f32) ys_f32[oidx] = hnew[e];
    }
    __syncthreads();                          // h_new visible for next step

    if (t + 1 < T_) {                         // hide serial-step xp latency
#pragma unroll
      for (int e = 0; e < 8; ++e) {
        const int m = e + hi * 8;
        __builtin_prefetch(
            xp + ((long long)(b0 + m) * T_ + (t + 1)) * G3_ + ucol, 0, 0);
      }
    }
  }

#pragma unroll
  for (int e = 0; e < 8; ++e) {
    const int m = e + hi * 8;
    h_final[(long long)(b0 + m) * U_ + (ub + nlo)] = hnew[e];
  }
}

// ---------------------------------------------------------------------------
extern "C" void kernel_launch(void* const* d_in, const int* in_sizes, int n_in,
                              void* d_out, int out_size, void* d_ws, size_t ws_size,
                              hipStream_t stream) {
  const float* x  = (const float*)d_in[0];
  const float* W0 = (const float*)d_in[1];
  const float* R0 = (const float*)d_in[2];
  const float* b0 = (const float*)d_in[3];
  const float* W1 = (const float*)d_in[4];
  const float* R1 = (const float*)d_in[5];
  const float* b1 = (const float*)d_in[6];
  float* out = (float*)d_out;

  char* ws = (char*)d_ws;
  size_t off = 0;
  auto wsalloc = [&](size_t bytes) -> void* {
    void* p = ws + off;
    off += (bytes + 255) & ~(size_t)255;
    return p;
  };

  unsigned short* x_bf  = (unsigned short*)wsalloc((size_t)B_ * T_ * IN_ * 2);
  unsigned short* y0_bf = (unsigned short*)wsalloc((size_t)B_ * T_ * U_ * 2);
  unsigned short* Wt0   = (unsigned short*)wsalloc((size_t)G3_ * IN_ * 2);
  unsigned short* Rt0   = (unsigned short*)wsalloc((size_t)G3_ * U_ * 2);
  unsigned short* Wt1   = (unsigned short*)wsalloc((size_t)G3_ * U_ * 2);
  unsigned short* Rt1   = (unsigned short*)wsalloc((size_t)G3_ * U_ * 2);
  float*          xp    = (float*)wsalloc((size_t)B_ * T_ * G3_ * 4);  // reused per layer

  // --- precision conversion / weight transposition (one-time, tiny) ---
  k_f32_to_bf16<<<2048, 256, 0, stream>>>(x, x_bf, (long long)B_ * T_ * IN_);
  k_transpose_bf16<<<1024, 256, 0, stream>>>(W0, Wt0, IN_, G3_);
  k_transpose_bf16<<<1024, 256, 0, stream>>>(R0, Rt0, U_, G3_);
  k_transpose_bf16<<<1024, 256, 0, stream>>>(W1, Wt1, U_, G3_);
  k_transpose_bf16<<<1024, 256, 0, stream>>>(R1, Rt1, U_, G3_);

  dim3 ggemm(G3_ / 64, (B_ * T_) / 128);
  float* h0_out = out + (long long)B_ * T_ * U_;
  float* h1_out = h0_out + (long long)B_ * U_;

  // Layer 0  (K = IN = 512)
  k_gemm_bf16<IN_><<<ggemm, 256, 0, stream>>>(x_bf, Wt0, b0, xp, B_ * T_, G3_);
  k_gru_recur<<<4, 1024, 0, stream>>>(Rt0, b0 + G3_, xp, y0_bf, nullptr, h0_out);
  // Layer 1  (K = U = 512)
  k_gemm_bf16<U_><<<ggemm, 256, 0, stream>>>(y0_bf, Wt1, b1, xp, B_ * T_, G3_);
  k_gru_recur<<<4, 1024, 0, stream>>>(Rt1, b1 + G3_, xp, nullptr, out, h1_out);
}
